// Aten_Encoder_48782238548149
// MI455X (gfx1250) — matrix operand order repaired and checked by reading, not compiled, not run on hardware
//
#include <hip/hip_runtime.h>
#include <hip/hip_bf16.h>

// Problem constants (match reference)
#define BB      8
#define SSQ     512
#define DDIM    1024
#define HHEADS  16
#define NLAYER  6
#define DFFN    512
#define DKH     64          // DDIM / HHEADS
#define LN_EPS  1e-6f
#define SC_LD   516         // padded LDS row stride (floats): 516%64==4 -> conflict-free rows

typedef __attribute__((ext_vector_type(16))) _Float16 v16h;
typedef __attribute__((ext_vector_type(8)))  _Float16 v8h;
typedef __attribute__((ext_vector_type(4)))  _Float16 v4h;
typedef __attribute__((ext_vector_type(8)))  float    v8f;

// ---------------------------------------------------------------------------
// WMMA fragment loaders (CDNA5 wave32, v_wmma_f32_16x16x32_f16 layouts §7.12.2)
// All global fragment loads are contiguous 16B chunks -> global_load_b128.
// ---------------------------------------------------------------------------

// A-matrix 16x32 f16 tile at (row0,k0) of a row-major matrix, lda in halves.
__device__ __forceinline__ v16h load_a_frag(const _Float16* __restrict__ A,
                                            int lda, int lane) {
  int m  = lane & 15;
  int kh = lane >> 4;
  const _Float16* p = A + (size_t)m * lda + kh * 8;
  v16h a;
#pragma unroll
  for (int t = 0; t < 8; ++t) { a[t] = p[t]; a[8 + t] = p[16 + t]; }
  return a;
}

// Same, fp32 source converted in registers (LDS attn tile -> A fragment).
__device__ __forceinline__ v16h load_a_frag_f32(const float* A, int lda, int lane) {
  int m  = lane & 15;
  int kh = lane >> 4;
  const float* p = A + (size_t)m * lda + kh * 8;
  v16h a;
#pragma unroll
  for (int t = 0; t < 8; ++t) {
    a[t]     = (_Float16)p[t];
    a[8 + t] = (_Float16)p[16 + t];
  }
  return a;
}

// B-matrix 32x16 tile where B[k][n] = Bt[n][k], Bt row-major (ldt halves).
// Caller pre-offsets Bt to (row=n0, col=k0). Two 8-half runs per lane (k=8h..8h+7, 16+8h..16+8h+7).
__device__ __forceinline__ v16h load_b_frag_t(const _Float16* __restrict__ Bt,
                                              int ldt, int lane) {
  int n    = lane & 15;
  int koff = (lane >> 4) * 8;
  const _Float16* p = Bt + (size_t)n * ldt + koff;
  v16h b;
#pragma unroll
  for (int t = 0; t < 8; ++t) { b[t] = p[t]; b[8 + t] = p[16 + t]; }
  return b;
}

__device__ __forceinline__ v8f wmma_f16(v16h a, v16h b, v8f c) {
  return __builtin_amdgcn_wmma_f32_16x16x32_f16(false, a, false, b,
                                                (short)0, c, false, false);
}

// ---------------------------------------------------------------------------
// Elementwise helpers (128-bit vectorized)
// ---------------------------------------------------------------------------
__global__ void copy_f32_kernel(const float* __restrict__ src,
                                float* __restrict__ dst, size_t n4) {
  size_t i = (size_t)blockIdx.x * blockDim.x + threadIdx.x;
  size_t stride = (size_t)gridDim.x * blockDim.x;
  const float4* s4 = (const float4*)src;
  float4* d4 = (float4*)dst;
  for (; i < n4; i += stride) d4[i] = s4[i];
}

// dst (f32) and dst16 (f16) both get a copy of src.
__global__ void copy_cvt_kernel(const float* __restrict__ src,
                                float* __restrict__ dst,
                                _Float16* __restrict__ dst16, size_t n4) {
  size_t i = (size_t)blockIdx.x * blockDim.x + threadIdx.x;
  size_t stride = (size_t)gridDim.x * blockDim.x;
  const float4* s4 = (const float4*)src;
  float4* d4 = (float4*)dst;
  v4h* d16 = (v4h*)dst16;
  for (; i < n4; i += stride) {
    float4 v = s4[i];
    d4[i] = v;
    v4h h;
    h[0] = (_Float16)v.x; h[1] = (_Float16)v.y;
    h[2] = (_Float16)v.z; h[3] = (_Float16)v.w;
    d16[i] = h;
  }
}

// ---------------------------------------------------------------------------
// Weight transpose+convert: Wt[l][n][k] = (f16) W[l][k][n]
// block (32,8), LDS-tiled for coalescing; grid (N/32, K/32, NLAYER)
// ---------------------------------------------------------------------------
__global__ void transpose_cvt_kernel(const float* __restrict__ W,
                                     _Float16* __restrict__ Wt,
                                     int K, int N) {
  __shared__ _Float16 tile[32][33];
  size_t loff = (size_t)blockIdx.z * K * N;
  const float* Wl = W + loff;
  _Float16* WtL = Wt + loff;
  int n0 = blockIdx.x * 32, k0 = blockIdx.y * 32;
  int tx = threadIdx.x, ty = threadIdx.y;
#pragma unroll
  for (int i = 0; i < 32; i += 8)
    tile[ty + i][tx] = (_Float16)Wl[(size_t)(k0 + ty + i) * N + n0 + tx];
  __syncthreads();
#pragma unroll
  for (int i = 0; i < 32; i += 8)
    WtL[(size_t)(n0 + ty + i) * K + k0 + tx] = tile[tx][ty + i];
}

// ---------------------------------------------------------------------------
// WMMA GEMM: out = act(A[M,K] @ Wt^T + bias),  Wt is [N,K] f16 (pre-transposed).
// Each wave computes a 32(M) x 64(N) tile: 2 A-frags x 4 B-frags = 8 WMMAs
// per k-step. block = 128 (4 waves over N strips); grid = (N/256, M/32).
// OUT_MODE: 0 = f32 [M,N], 1 = f16 [M,N], 2 = f16 transposed [N,M]
// ---------------------------------------------------------------------------
template <bool RELU, int OUT_MODE>
__global__ void gemm_tn_kernel(const _Float16* __restrict__ A,
                               const _Float16* __restrict__ Wt,
                               const float* __restrict__ bias,
                               void* __restrict__ out,
                               int M, int N, int K) {
  int lane = threadIdx.x & 31;
  int wave = threadIdx.x >> 5;
  int row0 = blockIdx.y * 32;
  int col0 = (blockIdx.x * 4 + wave) * 64;
  if (row0 >= M || col0 >= N) return;

  const _Float16* A0 = A + (size_t)row0 * K;
  const _Float16* A1 = A + (size_t)(row0 + 16) * K;

  v8f c[2][4] = {};
  for (int k0 = 0; k0 < K; k0 += 32) {
    if (k0 + 32 < K) {
      __builtin_prefetch(A0 + k0 + 32, 0, 1);
      __builtin_prefetch(A1 + k0 + 32, 0, 1);
    }
    v16h a0 = load_a_frag(A0 + k0, K, lane);
    v16h a1 = load_a_frag(A1 + k0, K, lane);
#pragma unroll
    for (int j = 0; j < 4; ++j) {
      v16h bf = load_b_frag_t(Wt + (size_t)(col0 + j * 16) * K + k0, K, lane);
      c[0][j] = wmma_f16(a0, bf, c[0][j]);
      c[1][j] = wmma_f16(a1, bf, c[1][j]);
    }
  }

  int n = lane & 15, rbase = (lane >> 4) * 8;
#pragma unroll
  for (int i = 0; i < 2; ++i) {
#pragma unroll
    for (int j = 0; j < 4; ++j) {
      int col = col0 + j * 16 + n;
      float bv = bias ? bias[col] : 0.0f;
      if (OUT_MODE == 2) {
        v8h pk;
#pragma unroll
        for (int r = 0; r < 8; ++r) {
          float v = c[i][j][r] + bv;
          if (RELU) v = fmaxf(v, 0.0f);
          pk[r] = (_Float16)v;
        }
        *(v8h*)((_Float16*)out + (size_t)col * M + row0 + i * 16 + rbase) = pk;
      } else {
#pragma unroll
        for (int r = 0; r < 8; ++r) {
          float v = c[i][j][r] + bv;
          if (RELU) v = fmaxf(v, 0.0f);
          size_t idx = (size_t)(row0 + i * 16 + rbase + r) * N + col;
          if (OUT_MODE == 1) ((_Float16*)out)[idx] = (_Float16)v;
          else               ((float*)out)[idx]    = v;
        }
      }
    }
  }
}

// ---------------------------------------------------------------------------
// Fused attention: scores (WMMA) -> softmax (LDS) -> attn-prob write (f32,
// coalesced) -> context = attn @ V (WMMA, A from LDS).
// One block per (q-tile of 16, b*H). block = 128 (4 waves).
//   scores phase: wave w owns keys [w*128, w*128+128) -> 8 accumulators.
//   softmax     : thread t -> (row = t>>3, 64-col segment = t&7) in LDS.
//   AV phase    : wave w owns dk-tile w (16 of DK=64); A-frags re-read from LDS.
// vT layout: vT[(h*DK+dk)*NTOK + b*S + s]  (from transposed V projection).
// ---------------------------------------------------------------------------
__global__ void fused_attn_kernel(const _Float16* __restrict__ q16,
                                  const _Float16* __restrict__ k16,
                                  const _Float16* __restrict__ vT,
                                  float* __restrict__ attnL,
                                  _Float16* __restrict__ o16, float scale) {
  __shared__ float sc[16 * SC_LD];
  __shared__ float redA[16][8];
  __shared__ float redB[16][8];
  const int NTOK = BB * SSQ;

  int t = threadIdx.x;
  int lane = t & 31, wave = t >> 5;
  int bh = blockIdx.y;
  int b = bh / HHEADS, h = bh % HHEADS;
  int q0 = blockIdx.x * 16;

  const _Float16* Q  = q16 + ((size_t)b * SSQ + q0) * DDIM + (size_t)h * DKH;
  const _Float16* Km = k16 + (size_t)b * SSQ * DDIM + (size_t)h * DKH;

  // ---- scores: 16 q x 128 keys per wave ----
  int key0 = wave * 128;
  {
    v8f c[8] = {};
#pragma unroll
    for (int k0 = 0; k0 < DKH; k0 += 32) {
      v16h a = load_a_frag(Q + k0, DDIM, lane);
#pragma unroll
      for (int j = 0; j < 8; ++j) {
        v16h bf = load_b_frag_t(Km + (size_t)(key0 + j * 16) * DDIM + k0,
                                DDIM, lane);
        c[j] = wmma_f16(a, bf, c[j]);
      }
    }
    int n = lane & 15, rbase = (lane >> 4) * 8;
#pragma unroll
    for (int j = 0; j < 8; ++j)
#pragma unroll
      for (int r = 0; r < 8; ++r)
        sc[(rbase + r) * SC_LD + key0 + j * 16 + n] = c[j][r] * scale;
  }
  __syncthreads();

  // ---- softmax over each of the 16 rows (512 entries) ----
  {
    int row = t >> 3, seg = t & 7;
    float4* rp = (float4*)(sc + row * SC_LD + seg * 64);
    float m = -3.4e38f;
#pragma unroll
    for (int i = 0; i < 16; ++i) {
      float4 v = rp[i];
      m = fmaxf(m, fmaxf(fmaxf(v.x, v.y), fmaxf(v.z, v.w)));
    }
    redA[row][seg] = m;
    __syncthreads();
#pragma unroll
    for (int i = 0; i < 8; ++i) m = fmaxf(m, redA[row][i]);
    float s = 0.0f;
#pragma unroll
    for (int i = 0; i < 16; ++i) {
      float4 v = rp[i];
      v.x = __expf(v.x - m); v.y = __expf(v.y - m);
      v.z = __expf(v.z - m); v.w = __expf(v.w - m);
      s += v.x + v.y + v.z + v.w;
      rp[i] = v;
    }
    redB[row][seg] = s;
    __syncthreads();
    s = 0.0f;
#pragma unroll
    for (int i = 0; i < 8; ++i) s += redB[row][i];
    float inv = 1.0f / s;
#pragma unroll
    for (int i = 0; i < 16; ++i) {
      float4 v = rp[i];
      v.x *= inv; v.y *= inv; v.z *= inv; v.w *= inv;
      rp[i] = v;
    }
  }
  __syncthreads();

  // ---- write attention probabilities (coalesced float4 sweep) ----
  {
    float* out = attnL + ((size_t)b * NLAYER * HHEADS + h) * (size_t)SSQ * SSQ +
                 (size_t)q0 * SSQ;
#pragma unroll
    for (int k = 0; k < 16; ++k) {
      int idx  = t + k * 128;        // 0..2047 float4 slots (16x512 floats)
      int rrow = idx >> 7;           // 128 float4 per row
      int c4   = idx & 127;
      float4 v = *(float4*)(sc + rrow * SC_LD + c4 * 4);
      ((float4*)out)[(size_t)rrow * 128 + c4] = v;
    }
  }

  // ---- context = attn @ V : wave w -> dk-tile w ----
  {
    const _Float16* Vb = vT + (size_t)(h * DKH + wave * 16) * NTOK +
                         (size_t)b * SSQ;
    v8f acc = {};
    for (int k0 = 0; k0 < SSQ; k0 += 32) {
      v16h a = load_a_frag_f32(sc + k0, SC_LD, lane);   // LDS -> ds loads
      v16h bf = load_b_frag_t(Vb + k0, NTOK, lane);
      acc = wmma_f16(a, bf, acc);
    }
    int n = lane & 15, rbase = (lane >> 4) * 8;
    _Float16* out = o16 + ((size_t)b * SSQ + q0) * DDIM + h * DKH + wave * 16;
#pragma unroll
    for (int r = 0; r < 8; ++r)
      out[(size_t)(rbase + r) * DDIM + n] = (_Float16)acc[r];
  }
}

// ---------------------------------------------------------------------------
// x = LN(x + y) over D=1024; also emits f16 copy of the result.
// 256 threads/row, 128-bit accesses.
// ---------------------------------------------------------------------------
__global__ void add_ln_kernel(float* __restrict__ x, const float* __restrict__ y,
                              const float* __restrict__ alpha,
                              const float* __restrict__ beta,
                              _Float16* __restrict__ x16out) {
  size_t row = blockIdx.x;
  float4* px = (float4*)(x + row * DDIM);
  const float4* py = (const float4*)(y + row * DDIM);
  int t = threadIdx.x;

  __shared__ float red[8];
  float4 a = px[t], b = py[t];
  float4 v = make_float4(a.x + b.x, a.y + b.y, a.z + b.z, a.w + b.w);
  float s = v.x + v.y + v.z + v.w;
#pragma unroll
  for (int off = 16; off; off >>= 1) s += __shfl_xor(s, off);
  if ((t & 31) == 0) red[t >> 5] = s;
  __syncthreads();
  s = 0.0f;
#pragma unroll
  for (int i = 0; i < 8; ++i) s += red[i];
  float mean = s * (1.0f / DDIM);
  __syncthreads();

  float dx = v.x - mean, dy = v.y - mean, dz = v.z - mean, dw = v.w - mean;
  float vs = dx * dx + dy * dy + dz * dz + dw * dw;
#pragma unroll
  for (int off = 16; off; off >>= 1) vs += __shfl_xor(vs, off);
  if ((t & 31) == 0) red[t >> 5] = vs;
  __syncthreads();
  vs = 0.0f;
#pragma unroll
  for (int i = 0; i < 8; ++i) vs += red[i];
  float inv = rsqrtf(vs * (1.0f / DDIM) + LN_EPS);

  const float4* al = (const float4*)alpha;
  const float4* be = (const float4*)beta;
  float4 av = al[t], bv = be[t];
  float4 o = make_float4(av.x * (dx * inv) + bv.x,
                         av.y * (dy * inv) + bv.y,
                         av.z * (dz * inv) + bv.z,
                         av.w * (dw * inv) + bv.w);
  px[t] = o;
  v4h hx;
  hx[0] = (_Float16)o.x; hx[1] = (_Float16)o.y;
  hx[2] = (_Float16)o.z; hx[3] = (_Float16)o.w;
  ((v4h*)(x16out + row * DDIM))[t] = hx;
}

// ---------------------------------------------------------------------------
// Host-side orchestration
// ---------------------------------------------------------------------------
extern "C" void kernel_launch(void* const* d_in, const int* in_sizes, int n_in,
                              void* d_out, int out_size, void* d_ws, size_t ws_size,
                              hipStream_t stream) {
  (void)in_sizes; (void)n_in; (void)out_size; (void)ws_size;

  const float* src  = (const float*)d_in[0];
  const float* Wq   = (const float*)d_in[1];
  const float* bq   = (const float*)d_in[2];
  const float* Wk   = (const float*)d_in[3];
  const float* bk   = (const float*)d_in[4];
  const float* Wv   = (const float*)d_in[5];
  const float* bv   = (const float*)d_in[6];
  const float* Wo   = (const float*)d_in[7];
  const float* bo   = (const float*)d_in[8];
  const float* ln1a = (const float*)d_in[9];
  const float* ln1b = (const float*)d_in[10];
  const float* ln2a = (const float*)d_in[11];
  const float* ln2b = (const float*)d_in[12];
  const float* W1   = (const float*)d_in[13];
  const float* b1   = (const float*)d_in[14];
  const float* W2   = (const float*)d_in[15];
  const float* b2   = (const float*)d_in[16];

  const size_t NTOK = (size_t)BB * SSQ;               // 4096
  const size_t NX   = NTOK * DDIM;                    // x elements
  const size_t WSQ  = (size_t)NLAYER * DDIM * DDIM;   // Wq/Wk/Wv/Wo elems
  const size_t WFF  = (size_t)NLAYER * DDIM * DFFN;   // W1/W2 elems

  char* cur = (char*)d_ws;
  auto alloc = [&](size_t bytes) -> void* {
    void* p = (void*)cur;
    cur += (bytes + 255) & ~(size_t)255;
    return p;
  };
  // transposed f16 weights: [NL][N][K]
  _Float16* wqT = (_Float16*)alloc(WSQ * 2);
  _Float16* wkT = (_Float16*)alloc(WSQ * 2);
  _Float16* wvT = (_Float16*)alloc(WSQ * 2);
  _Float16* woT = (_Float16*)alloc(WSQ * 2);
  _Float16* w1T = (_Float16*)alloc(WFF * 2);   // [NL][DFF][D]
  _Float16* w2T = (_Float16*)alloc(WFF * 2);   // [NL][D][DFF]
  _Float16* x16 = (_Float16*)alloc(NX * 2);
  _Float16* q16 = (_Float16*)alloc(NX * 2);
  _Float16* k16 = (_Float16*)alloc(NX * 2);
  _Float16* vT  = (_Float16*)alloc(NX * 2);    // transposed V: [D][NTOK]
  _Float16* o16 = (_Float16*)alloc(NX * 2);
  _Float16* h116= (_Float16*)alloc(NTOK * DFFN * 2);
  float*    xbuf= (float*)alloc(NX * 4);
  float*    t32 = (float*)alloc(NX * 4);

  float* out_x    = (float*)d_out;
  float* out_attn = out_x + NX;   // [B, NL, H, S, S]

  const dim3 blkT(32, 8), blk128(128), blk256(256);

  // Transpose+convert all weights once per launch (deterministic).
  {
    dim3 gQ(DDIM / 32, DDIM / 32, NLAYER);
    transpose_cvt_kernel<<<gQ, blkT, 0, stream>>>(Wq, wqT, DDIM, DDIM);
    transpose_cvt_kernel<<<gQ, blkT, 0, stream>>>(Wk, wkT, DDIM, DDIM);
    transpose_cvt_kernel<<<gQ, blkT, 0, stream>>>(Wv, wvT, DDIM, DDIM);
    transpose_cvt_kernel<<<gQ, blkT, 0, stream>>>(Wo, woT, DDIM, DDIM);
    dim3 g1(DFFN / 32, DDIM / 32, NLAYER);   // W1: K=D, N=DFF
    transpose_cvt_kernel<<<g1, blkT, 0, stream>>>(W1, w1T, DDIM, DFFN);
    dim3 g2(DDIM / 32, DFFN / 32, NLAYER);   // W2: K=DFF, N=D
    transpose_cvt_kernel<<<g2, blkT, 0, stream>>>(W2, w2T, DFFN, DDIM);
  }

  // x = src (f32 + f16 copies)
  copy_cvt_kernel<<<4096, 256, 0, stream>>>(src, xbuf, x16, NX / 4);

  const float scale = 0.125f;   // 1/sqrt(DK=64)

  for (int i = 0; i < NLAYER; ++i) {
    const _Float16* wqL = wqT + (size_t)i * DDIM * DDIM;
    const _Float16* wkL = wkT + (size_t)i * DDIM * DDIM;
    const _Float16* wvL = wvT + (size_t)i * DDIM * DDIM;
    const _Float16* woL = woT + (size_t)i * DDIM * DDIM;
    const _Float16* w1L = w1T + (size_t)i * DDIM * DFFN;
    const _Float16* w2L = w2T + (size_t)i * DFFN * DDIM;

    // Q, K projections (f16 [M,N]); V projection written transposed [N,M]
    dim3 gP(DDIM / 256, (unsigned)(NTOK / 32));
    gemm_tn_kernel<false, 1><<<gP, blk128, 0, stream>>>(
        x16, wqL, bq + (size_t)i * DDIM, q16, (int)NTOK, DDIM, DDIM);
    gemm_tn_kernel<false, 1><<<gP, blk128, 0, stream>>>(
        x16, wkL, bk + (size_t)i * DDIM, k16, (int)NTOK, DDIM, DDIM);
    gemm_tn_kernel<false, 2><<<gP, blk128, 0, stream>>>(
        x16, wvL, bv + (size_t)i * DDIM, vT, (int)NTOK, DDIM, DDIM);

    // fused scores + softmax + attn write + attn@V
    float* attnL = out_attn + (size_t)i * HHEADS * SSQ * SSQ;
    dim3 gA(SSQ / 16, BB * HHEADS);
    fused_attn_kernel<<<gA, blk128, 0, stream>>>(q16, k16, vT, attnL, o16, scale);

    // output projection (f32 out) + residual + LN1 (emits f16 x)
    gemm_tn_kernel<false, 0><<<gP, blk128, 0, stream>>>(
        o16, woL, bo + (size_t)i * DDIM, t32, (int)NTOK, DDIM, DDIM);
    add_ln_kernel<<<(unsigned)NTOK, blk256, 0, stream>>>(
        xbuf, t32, ln1a + (size_t)i * DDIM, ln1b + (size_t)i * DDIM, x16);

    // FFN
    dim3 gF1(DFFN / 256, (unsigned)(NTOK / 32));
    gemm_tn_kernel<true, 1><<<gF1, blk128, 0, stream>>>(
        x16, w1L, b1 + (size_t)i * DFFN, h116, (int)NTOK, DFFN, DDIM);
    dim3 gF2(DDIM / 256, (unsigned)(NTOK / 32));
    gemm_tn_kernel<false, 0><<<gF2, blk128, 0, stream>>>(
        h116, w2L, b2 + (size_t)i * DDIM, t32, (int)NTOK, DDIM, DFFN);
    add_ln_kernel<<<(unsigned)NTOK, blk256, 0, stream>>>(
        xbuf, t32, ln2a + (size_t)i * DDIM, ln2b + (size_t)i * DDIM, x16);
  }

  // write final x
  copy_f32_kernel<<<4096, 256, 0, stream>>>(xbuf, out_x, NX / 4);
}
